// GraphNet_83958020702804
// MI455X (gfx1250) — matrix operand order repaired and checked by reading, not compiled, not run on hardware
//
#include <hip/hip_runtime.h>

// ---------------- types for WMMA ----------------
typedef __attribute__((ext_vector_type(16))) __bf16 v16bf;
typedef __attribute__((ext_vector_type(4)))  __bf16 v4bf;
typedef __attribute__((ext_vector_type(8)))  float  v8f;

#define NN   100000
#define E2   1600000      // 2E edges
#define HALF_TILES 50000  // E/16 per half
#define DD   100
#define KTOT 320          // 3*100 padded to 10*32
#define ASTR 328          // LDS row stride (bank-conflict padding)
#define NPAD 112          // 7 tiles * 16
#define K2   128          // MLP K padded to 4*32
#define MSTR 136          // MLP LDS row stride

// WlT[112][320], Wl2T[112][320], W1T[112][128], W2T[112][128] in ws (bf16)
#define OFF_WL2  (NPAD * KTOT)
#define OFF_W1   (2 * NPAD * KTOT)
#define OFF_W2   (2 * NPAD * KTOT + NPAD * K2)

// A fragment: 16-bit A 16x32 layout. p -> A[m] + sel + kc*32 (LDS, bf16 elems).
// elems 0..7 = K base+0..7, elems 8..15 = K base+16..23.
__device__ inline v16bf load_frag_a(const __bf16* p) {
    v16bf r;
    uint4* u = reinterpret_cast<uint4*>(&r);
    u[0] = *reinterpret_cast<const uint4*>(p);
    u[1] = *reinterpret_cast<const uint4*>(p + 16);
    return r;
}

// B fragment: lane n<16 holds K=kc*32+0..7,16..23 for column n; lanes 16..31 hold K=+8..15,24..31.
__device__ inline v16bf load_frag_b(const __bf16* p) {
    v16bf r;
    uint4* u = reinterpret_cast<uint4*>(&r);
    u[0] = reinterpret_cast<const uint4*>(p)[0];
    u[1] = reinterpret_cast<const uint4*>(p + 16)[0];
    return r;
}

__device__ inline v4bf cvt4(float4 v) {
    v4bf w = { (__bf16)v.x, (__bf16)v.y, (__bf16)v.z, (__bf16)v.w };
    return w;
}

// ---------------- kernel 0: weight prep (fp32 -> transposed padded bf16) ----------------
__global__ void prep_kernel(const float* __restrict__ Wl,  const float* __restrict__ Wl2,
                            const float* __restrict__ W1,  const float* __restrict__ W2,
                            __bf16* __restrict__ ws) {
    int i = blockIdx.x * blockDim.x + threadIdx.x;
    int stride = gridDim.x * blockDim.x;
    for (int idx = i; idx < NPAD * KTOT; idx += stride) {
        int nrow = idx / KTOT, k = idx - nrow * KTOT;
        bool ok = (nrow < DD) && (k < 3 * DD);
        ws[idx]           = (__bf16)(ok ? Wl [k * DD + nrow] : 0.0f);
        ws[OFF_WL2 + idx] = (__bf16)(ok ? Wl2[k * DD + nrow] : 0.0f);
    }
    for (int idx = i; idx < NPAD * K2; idx += stride) {
        int nrow = idx / K2, k = idx - nrow * K2;
        bool ok = (nrow < DD) && (k < DD);
        ws[OFF_W1 + idx] = (__bf16)(ok ? W1[k * DD + nrow] : 0.0f);
        ws[OFF_W2 + idx] = (__bf16)(ok ? W2[k * DD + nrow] : 0.0f);
    }
}

// ---------------- kernel 1: out = (1+eps)*x  (out doubles as aggregation buffer) ----------
__global__ void init_kernel(const float* __restrict__ x, const float* __restrict__ eps,
                            float* __restrict__ out) {
    const float s = 1.0f + eps[0];
    size_t i = (size_t)blockIdx.x * blockDim.x + threadIdx.x;
    size_t stride = (size_t)gridDim.x * blockDim.x;
    const float4* __restrict__ x4 = (const float4*)x;
    float4* __restrict__ o4 = (float4*)out;
    for (size_t idx = i; idx < (size_t)NN * DD / 4; idx += stride) {
        float4 v = x4[idx];
        v.x *= s; v.y *= s; v.z *= s; v.w *= s;
        o4[idx] = v;
    }
}

// ---------------- kernel 2: edge messages + atomic scatter-add ----------------------------
// 2048 blocks x 224 threads (7 waves). Blocks [0,1024) -> half 0 (Wl), [1024,2048) -> half 1.
// Double-buffered LDS gather: 1 barrier per 16-edge tile, next tile's gather overlaps WMMA.
__global__ __launch_bounds__(224) void edge_kernel(
    const float* __restrict__ x, const int* __restrict__ ei,
    const float* __restrict__ ea, const __bf16* __restrict__ wsW,
    const float* __restrict__ bl, const float* __restrict__ bl2,
    float* __restrict__ out) {
    __shared__ __bf16 ldsA[2][16 * ASTR];

    const int tid  = threadIdx.x;
    const int lane = tid & 31;
    const int wv   = tid >> 5;                 // 0..6 : N-tile owned by this wave
    const int hsel = (blockIdx.x >= 1024) ? 1 : 0;
    const int bidx = blockIdx.x & 1023;

    const int* __restrict__ srcI = ei;         // edge_index[0]
    const int* __restrict__ dstI = ei + E2;    // edge_index[1]
    const __bf16* __restrict__ WT = wsW + (size_t)hsel * (NPAD * KTOT);
    const float* __restrict__ bias_p = hsel ? bl2 : bl;

    // per-wave B fragments held in registers for the whole tile loop (L2-resident weights)
    const int nloc = lane & 15;
    const int ncol = wv * 16 + nloc;
    const int kb   = (lane < 16) ? 0 : 8;
    v16bf Bf[10];
    {
        const __bf16* base = WT + (size_t)ncol * KTOT + kb;
#pragma unroll
        for (int kc = 0; kc < 10; ++kc) Bf[kc] = load_frag_b(base + kc * 32);
    }
    const float bias  = (ncol < DD) ? bias_p[ncol] : 0.0f;
    const int   mbase = (lane >> 4) * 8;
    const int   sel   = (lane < 16) ? 0 : 8;

    // vectorized gather of one 16-edge tile into a buffer: 1200 float4 quads
    auto fillA = [&](int t, __bf16* dst) {
        const int eb = (hsel * HALF_TILES + t) * 16;
        for (int idx = tid; idx < 3 * 16 * 25; idx += 224) {
            int seg = idx / 400;               // 0: f_a, 1: edge_attr, 2: f_b
            int rem = idx - seg * 400;
            int row = rem / 25;
            int q   = rem - row * 25;
            int e   = eb + row;
            const float4* s4;
            if (seg == 1) {
                s4 = (const float4*)(ea + (size_t)e * DD);
            } else {
                int node;
                if (seg == 0) node = hsel ? srcI[e] : dstI[e];
                else          node = hsel ? dstI[e] : srcI[e];
                s4 = (const float4*)(x + (size_t)node * DD);
            }
            *(v4bf*)(&dst[row * ASTR + seg * DD + q * 4]) = cvt4(s4[q]);
        }
    };

    // one-time zero of the constant K-padding cols [300,320) of both buffers
    {
        v4bf z = {};
        for (int idx = tid; idx < 2 * 16 * 5; idx += 224) {
            int b = idx / 80, rem = idx - b * 80;
            int row = rem / 5, q = rem - row * 5;
            *(v4bf*)(&ldsA[b][row * ASTR + 300 + q * 4]) = z;
        }
    }
    fillA(bidx, ldsA[0]);
    __syncthreads();

    int p = 0;
    for (int t = bidx; t < HALF_TILES; t += 1024) {
        const int tn = t + 1024;
        if (tn < HALF_TILES) fillA(tn, ldsA[p ^ 1]);   // overlap next gather with WMMA

        const int eb = (hsel * HALF_TILES + t) * 16;
        v8f acc = {};
        {
            const __bf16* arow = ldsA[p] + nloc * ASTR + sel;
#pragma unroll
            for (int kc = 0; kc < 10; ++kc) {
                v16bf Af = load_frag_a(arow + kc * 32);
                acc = __builtin_amdgcn_wmma_f32_16x16x32_bf16(
                    false, Af, false, Bf[kc], (short)0, acc, false, false);
            }
        }
        if (ncol < DD) {
#pragma unroll
            for (int r = 0; r < 8; ++r) {
                int e = eb + mbase + r;
                float v = acc[r] + bias;
                v = v > 0.0f ? v : 0.0f;
                int dn = dstI[e];
                atomicAdd(out + (size_t)dn * DD + ncol, v);  // coalesced across lanes 0..15
            }
        }
        __syncthreads();
        p ^= 1;
    }
}

// ---------------- kernel 3: fused node MLP, in place on out -------------------------------
__global__ __launch_bounds__(224) void mlp_kernel(
    const __bf16* __restrict__ W1T, const __bf16* __restrict__ W2T,
    const float* __restrict__ b1,   const float* __restrict__ b2,
    float* __restrict__ out) {
    __shared__ __bf16 ldsA[16 * MSTR];
    __shared__ __bf16 ldsH[16 * MSTR];

    const int tid  = threadIdx.x;
    const int lane = tid & 31;
    const int wv   = tid >> 5;
    const size_t grow0 = (size_t)blockIdx.x * 16;

    // vectorized load of 16 node rows (float4 -> packed bf16), plus zero padding
    for (int idx = tid; idx < 16 * 25; idx += 224) {
        int row = idx / 25, q = idx - row * 25;
        float4 v = *(const float4*)(out + (grow0 + row) * DD + q * 4);
        *(v4bf*)(&ldsA[row * MSTR + q * 4]) = cvt4(v);
    }
    {
        v4bf z = {};
        for (int idx = tid; idx < 16 * 9; idx += 224) {   // A cols [100,136)
            int row = idx / 9, q = idx - row * 9;
            *(v4bf*)(&ldsA[row * MSTR + DD + q * 4]) = z;
        }
        for (int idx = tid; idx < 16 * 6; idx += 224) {   // H cols [112,136)
            int row = idx / 6, q = idx - row * 6;
            *(v4bf*)(&ldsH[row * MSTR + NPAD + q * 4]) = z;
        }
    }
    __syncthreads();

    const int nloc  = lane & 15;
    const int ncol  = wv * 16 + nloc;
    const int kb    = (lane < 16) ? 0 : 8;
    const int sel   = (lane < 16) ? 0 : 8;
    const int mbase = (lane >> 4) * 8;

    // stage 1: h = relu(r @ W1 + b1)
    v8f acc = {};
#pragma unroll
    for (int kc = 0; kc < 4; ++kc) {
        v16bf Af = load_frag_a(ldsA + nloc * MSTR + sel + kc * 32);
        v16bf Bf = load_frag_b(W1T + (size_t)ncol * K2 + kb + kc * 32);
        acc = __builtin_amdgcn_wmma_f32_16x16x32_bf16(
            false, Af, false, Bf, (short)0, acc, false, false);
    }
    const float bias1 = (ncol < DD) ? b1[ncol] : 0.0f;
#pragma unroll
    for (int r = 0; r < 8; ++r) {
        float v = acc[r] + bias1;
        v = v > 0.0f ? v : 0.0f;
        ldsH[(mbase + r) * MSTR + ncol] = (__bf16)v;
    }
    __syncthreads();

    // stage 2: out = h @ W2 + b2 (in place; this block owns these 16 rows)
    v8f acc2 = {};
#pragma unroll
    for (int kc = 0; kc < 4; ++kc) {
        v16bf Af = load_frag_a(ldsH + nloc * MSTR + sel + kc * 32);
        v16bf Bf = load_frag_b(W2T + (size_t)ncol * K2 + kb + kc * 32);
        acc2 = __builtin_amdgcn_wmma_f32_16x16x32_bf16(
            false, Af, false, Bf, (short)0, acc2, false, false);
    }
    if (ncol < DD) {
        const float bias2 = b2[ncol];
#pragma unroll
        for (int r = 0; r < 8; ++r)
            out[(grow0 + mbase + r) * DD + ncol] = acc2[r] + bias2;
    }
}

// ---------------- launcher ----------------------------------------------------------------
extern "C" void kernel_launch(void* const* d_in, const int* in_sizes, int n_in,
                              void* d_out, int out_size, void* d_ws, size_t ws_size,
                              hipStream_t stream) {
    const float* x   = (const float*)d_in[0];
    const int*   ei  = (const int*)  d_in[1];
    const float* ea  = (const float*)d_in[2];
    const float* Wl  = (const float*)d_in[3];
    const float* bl  = (const float*)d_in[4];
    const float* Wl2 = (const float*)d_in[5];
    const float* bl2 = (const float*)d_in[6];
    const float* W1  = (const float*)d_in[7];
    const float* b1  = (const float*)d_in[8];
    const float* W2  = (const float*)d_in[9];
    const float* b2  = (const float*)d_in[10];
    const float* eps = (const float*)d_in[11];
    float* out = (float*)d_out;
    __bf16* ws = (__bf16*)d_ws;

    prep_kernel<<<64, 256, 0, stream>>>(Wl, Wl2, W1, W2, ws);
    init_kernel<<<2048, 256, 0, stream>>>(x, eps, out);
    edge_kernel<<<2048, 224, 0, stream>>>(x, ei, ea, ws, bl, bl2, out);
    mlp_kernel<<<NN / 16, 224, 0, stream>>>(ws + OFF_W1, ws + OFF_W2, b1, b2, out);
}
